// VCharge_14671608283193
// MI455X (gfx1250) — 8-bit matrix operand order repaired and checked by reading, compiled, not run on hardware
//
#include <hip/hip_runtime.h>
#include <stdint.h>

typedef __attribute__((ext_vector_type(8))) int v8i;

#define NATOMS 64
#define NT 39
#define WPB 8

__global__ __launch_bounds__(256) void vcharge_kernel(
    const int*   __restrict__ atoms,
    const float* __restrict__ A,
    const float* __restrict__ Q,
    const float* __restrict__ e,
    const float* __restrict__ s,
    const float* __restrict__ alpha,
    const float* __restrict__ betap,
    float*       __restrict__ out,
    int B)
{
    __shared__ float    Ptab[NT * NT];
    __shared__ uint32_t AfP[WPB][NATOMS * 16];
    __shared__ uint8_t  atInfo[WPB][NATOMS];
    __shared__ float    sinvA[WPB][NATOMS];
    __shared__ float    ecolP[WPB][2 * NATOMS];
    __shared__ float    red[WPB][64];
    __shared__ float    lamS[WPB];

    const int tid  = threadIdx.x;
    const int lane = tid & 31;
    const int wave = tid >> 5;
    const int half = lane >> 4;
    const int l15  = lane & 15;
    int b = blockIdx.x * WPB + wave;
    if (b >= B) b = B - 1;

    const float beta = betap[0];

    for (int t = tid; t < NT * NT; t += blockDim.x) {
        int ai = t / NT, aj = t % NT;
        float d  = e[aj] - e[ai];
        float p  = __powf(fabsf(d), beta);
        Ptab[t]  = (d < 0.f) ? -p : p;
    }

    uint32_t* af  = AfP[wave];
    uint8_t*  af8 = (uint8_t*)af;
    for (int k = lane; k < NATOMS * 16; k += 32) af[k] = 0u;

    const uint64_t AROM = (1ull << 6) | (1ull << 10) | (1ull << 19) | (1ull << 20) |
                          (1ull << 21) | (1ull << 22) | (1ull << 32);
    for (int j = lane; j < NATOMS; j += 32) {
        int t  = atoms[b * NATOMS + j];
        int ar = (int)((AROM >> (t & 63)) & 1ull);
        atInfo[wave][j] = (uint8_t)((t & 0x3f) | (ar << 7));
        sinvA[wave][j]  = 1.0f / s[t];
    }
    __syncthreads();

    const float4* A4 = (const float4*)(A + (size_t)b * NATOMS * NATOMS);
    for (int g = lane; g < NATOMS * NATOMS / 4; g += 32) {
        float4 v = A4[g];
        int i  = g >> 4;
        int j0 = (g & 15) * 4;
        float vv[4] = {v.x, v.y, v.z, v.w};
        #pragma unroll
        for (int t = 0; t < 4; ++t) {
            int j = j0 + t;
            int u = (int)vv[t];
            if (j > i && u != 0) {
                uint8_t ub = (uint8_t)u;
                af8[i * NATOMS + j] = ub;
                af8[j * NATOMS + i] = ub;
            }
        }
    }
    __syncthreads();

    v8i Bfrag[4];
    #pragma unroll
    for (int J = 0; J < 4; ++J) {
        const uint32_t* row = af + (16 * J + l15) * 16;
        #pragma unroll
        for (int d = 0; d < 8; ++d)
            Bfrag[J][d] = (int)row[8 * (d >> 2) + 4 * half + (d & 3)];
    }

    const float a0 = alpha[0], a1 = alpha[1], a2 = alpha[2], a3 = alpha[3], a4 = alpha[4];
    float acc[4] = {0.f, 0.f, 0.f, 0.f};

    #pragma unroll
    for (int I = 0; I < 4; ++I) {
        const uint32_t* row = af + (16 * I + l15) * 16;
        v8i Afrag;
        #pragma unroll
        for (int d = 0; d < 8; ++d)
            Afrag[d] = (int)row[8 * (d >> 2) + 4 * half + (d & 3)];

        #pragma unroll
        for (int J = 0; J < 4; ++J) {
            v8i c = {};
            c = __builtin_amdgcn_wmma_i32_16x16x64_iu8(false, Afrag, false, Bfrag[J],
                                                       c, false, false);
            const int j = 16 * J + l15;
            const uint8_t infoJ = atInfo[wave][j];
            const int   atj   = infoJ & 0x3f;
            const float aromj = (infoJ >> 7) ? 1.f : 0.f;
            #pragma unroll
            for (int v = 0; v < 8; ++v) {
                const int i = 16 * I + 8 * half + v;
                const uint8_t afij  = af8[i * NATOMS + j];
                const uint8_t infoI = atInfo[wave][i];
                const int   ati   = infoI & 0x3f;
                const float aromi = (infoI >> 7) ? 1.f : 0.f;
                const float one3  = (c[v] > 0) ? 1.f : 0.f;
                const float m1 = (afij == 1) ? 1.f : 0.f;
                const float m2 = (afij == 2) ? 1.f : 0.f;
                const float m3 = (afij == 3) ? 1.f : 0.f;
                const float marom = (afij != 0) ? aromi * aromj : 0.f;
                const float w = a0 * m1 + a1 * m2 + a2 * m3 + a3 * marom + a4 * one3;
                acc[J] += w * Ptab[ati * NT + atj];
            }
        }
    }

    #pragma unroll
    for (int J = 0; J < 4; ++J)
        ecolP[wave][half * NATOMS + 16 * J + l15] = acc[J];
    __syncthreads();

    float pes = 0.f, ps = 0.f;
    float ecolv[2], siv[2];
    #pragma unroll
    for (int t = 0; t < 2; ++t) {
        const int j = lane + 32 * t;
        const float ec = ecolP[wave][j] + ecolP[wave][NATOMS + j];
        const float sv = sinvA[wave][j];
        ecolv[t] = ec; siv[t] = sv;
        pes += ec * sv;
        ps  += sv;
    }
    red[wave][lane]      = pes;
    red[wave][32 + lane] = ps;
    __syncthreads();
    if (lane == 0) {
        float S1 = 0.f, S2 = 0.f;
        for (int k = 0; k < 32; ++k) { S1 += red[wave][k]; S2 += red[wave][32 + k]; }
        lamS[wave] = (Q[b] + S1) / S2;
    }
    __syncthreads();
    const float lam = lamS[wave];

    #pragma unroll
    for (int t = 0; t < 2; ++t) {
        const int j = lane + 32 * t;
        out[(size_t)b * NATOMS + j] = siv[t] * (lam - ecolv[t]);
    }
}

extern "C" void kernel_launch(void* const* d_in, const int* in_sizes, int n_in,
                              void* d_out, int out_size, void* d_ws, size_t ws_size,
                              hipStream_t stream) {
    const int*   atoms = (const int*)  d_in[0];
    const float* A     = (const float*)d_in[1];
    const float* Q     = (const float*)d_in[2];
    const float* e     = (const float*)d_in[3];
    const float* s     = (const float*)d_in[4];
    const float* alpha = (const float*)d_in[5];
    const float* beta  = (const float*)d_in[6];
    float* out = (float*)d_out;

    const int B = in_sizes[2];
    const int blocks = (B + WPB - 1) / WPB;
    vcharge_kernel<<<blocks, 256, 0, stream>>>(atoms, A, Q, e, s, alpha, beta, out, B);
}
